// MultiHeadAttention_67860483276894
// MI455X (gfx1250) — compile-verified, matrix operand order checked by reading, not run on hardware
//
#include <hip/hip_runtime.h>
#include <math.h>

// ---------------------------------------------------------------------------
// MHA for gfx1250 (CDNA5, wave32, WMMA + TDM).  B=4, L=2048, D=1024, H=16.
// fp32 in -> bf16 WMMA pipeline -> fp32 out.  K/V tiles staged by the
// Tensor Data Mover (double-buffered); softmax row reductions on DPP16.
// ---------------------------------------------------------------------------

typedef __attribute__((ext_vector_type(16))) __bf16 v16bf;
typedef __attribute__((ext_vector_type(8)))  __bf16 v8bf;
typedef __attribute__((ext_vector_type(8)))  float  v8f;
typedef __attribute__((ext_vector_type(4)))  float  v4f;
typedef __attribute__((ext_vector_type(4)))  unsigned int v4u;
typedef __attribute__((ext_vector_type(8)))  int    v8i;
typedef __attribute__((ext_vector_type(4)))  int    v4i;

#define BATCH   4
#define SEQ     2048
#define DMODEL  1024
#define NHEADS  16
#define DK      64
#define MROWS   (BATCH * SEQ)   // 8192

__device__ __forceinline__ v16bf cat8(v8bf lo, v8bf hi) {
    return __builtin_shufflevector(lo, hi, 0,1,2,3,4,5,6,7,8,9,10,11,12,13,14,15);
}

// Load a 16x32 bf16 A/B fragment from a row-major [rows][stride] bf16 buffer.
// Per-lane element order matches the CDNA5 WMMA VGPR layout:
// elems 0..7 = K(koff..koff+7), 8..15 = K(16+koff..23+koff).
__device__ __forceinline__ v16bf load_frag(const __bf16* p) {
    v8bf lo = *(const v8bf*)p;         // 16B -> ds_load_b128 / global_load_b128
    v8bf hi = *(const v8bf*)(p + 16);
    return cat8(lo, hi);
}

__device__ __forceinline__ v8f wmma_bf16(v16bf a, v16bf b, v8f c) {
    return __builtin_amdgcn_wmma_f32_16x16x32_bf16(
        /*neg_a=*/false, a, /*neg_b=*/false, b,
        /*c_mod=*/(short)0, c, /*reuse_a=*/false, /*reuse_b=*/false);
}

// ---------------------------------------------------------------------------
// Butterfly reductions across a 16-lane row entirely on the VALU (DPP16),
// no ds_bpermute.  Steps: xor1 (quad_perm 0xB1), xor2 (quad_perm 0x4E),
// then row_half_mirror (0x141) and row_mirror (0x140), which act as xor4 /
// xor8 because lane groups are already uniform after the previous steps.
// dpp_ctrl must be a literal constant -> template parameter.
// ---------------------------------------------------------------------------
template<int CTRL>
__device__ __forceinline__ float dpp_step(float v) {
    return __int_as_float(
        __builtin_amdgcn_update_dpp(0, __float_as_int(v), CTRL, 0xF, 0xF, true));
}
__device__ __forceinline__ float row16_max(float v) {
    v = fmaxf(v, dpp_step<0xB1>(v));    // quad_perm [1,0,3,2]  (xor 1)
    v = fmaxf(v, dpp_step<0x4E>(v));    // quad_perm [2,3,0,1]  (xor 2)
    v = fmaxf(v, dpp_step<0x141>(v));   // row_half_mirror      (xor 4)
    v = fmaxf(v, dpp_step<0x140>(v));   // row_mirror           (xor 8)
    return v;
}
__device__ __forceinline__ float row16_sum(float v) {
    v += dpp_step<0xB1>(v);
    v += dpp_step<0x4E>(v);
    v += dpp_step<0x141>(v);
    v += dpp_step<0x140>(v);
    return v;
}

// ---------------------------------------------------------------------------
// TDM: load a 64x64 bf16 tile (tile_dim1 x tile_dim0) from a 2D tensor into
// LDS, with hardware padding of 4 DWORDs (8 bf16) after every 32 DWORDs
// (64 bf16 = one tile row) -> LDS row stride of 72 elements.
// D# layout per CDNA5 ISA ch.8.  Toolchain uses the 6-arg builtin:
//   (uint32x4 g0, int32x8 g1, int32x4 g2, int32x4 g3, int32x8, i32 cpol)
// ---------------------------------------------------------------------------
__device__ __forceinline__ void tdm_load_tile64(unsigned int lds_addr,
                                                const void* gaddr,
                                                unsigned int tensor_d0,
                                                unsigned int tensor_d1,
                                                unsigned int stride_d0) {
    unsigned long long ga = (unsigned long long)(uintptr_t)gaddr;
    v4u g0;
    g0[0] = 1u;                                            // count=1 (valid)
    g0[1] = lds_addr;                                      // lds_addr[31:0]
    g0[2] = (unsigned int)ga;                              // global_addr[31:0]
    g0[3] = (unsigned int)((ga >> 32) & 0x1FFFFFFu)        // global_addr[56:32]
          | (2u << 30);                                    // type=2 (image)
    v8i g1;
    g1[0] = (1 << 16)            // data_size = 1 -> 2 bytes
          | (1 << 20)            // pad_enable
          | (4 << 22)            // pad_interval code 4 -> every 32 DWORDs
          | (3 << 25);           // pad_amount  code 3 -> 4 DWORDs (8 bf16)
    g1[1] = (int)((tensor_d0 & 0xFFFFu) << 16);            // tensor_dim0[15:0]
    g1[2] = (int)(((tensor_d0 >> 16) & 0xFFFFu)            // tensor_dim0[31:16]
          | ((tensor_d1 & 0xFFFFu) << 16));                // tensor_dim1[15:0]
    g1[3] = (int)(((tensor_d1 >> 16) & 0xFFFFu)            // tensor_dim1[31:16]
          | (64u << 16));                                  // tile_dim0 = 64
    g1[4] = 64;                                            // tile_dim1 = 64
    g1[5] = (int)stride_d0;                                // dim0_stride[31:0]
    g1[6] = 0;                                             // stride hi / dim1_stride
    g1[7] = 0;
    v4i z4 = {0, 0, 0, 0};                                 // 2D: groups 2/3 unused
    v8i z8 = {0, 0, 0, 0, 0, 0, 0, 0};
    __builtin_amdgcn_tensor_load_to_lds(g0, g1, z4, z4, z8, 0);
}

// ---------------------------------------------------------------------------
// Tiled GEMM:  Out[M,N] = A[M,K] * W[K,N] + bias[N]
//   block tile 128x64, kc=64, 8 waves, each wave 32x32 (2x2 WMMA tiles).
//   A_BF16=false : A is fp32 (x), converted to bf16 on the fly.
//   A_BF16=true  : A is bf16 (attention context).
//   OUT_MODE=0   : fp32 row-major [M,N]                  (final Wo GEMM)
//   OUT_MODE=1   : bf16 head layout [B,H,L,DK]           (Q/K projections)
//   OUT_MODE=2   : bf16 transposed head layout [B,H,DK,L] (V projection)
// ---------------------------------------------------------------------------
template<bool A_BF16, int OUT_MODE>
__global__ __launch_bounds__(256)
void gemm_wmma_kernel(const void* __restrict__ Aptr,
                      const float* __restrict__ W,
                      const float* __restrict__ bias,
                      void* __restrict__ Out,
                      int M, int N, int Kdim) {
    constexpr int LST = 72;                     // 64 + 8 pad (keeps 16B align)
    __shared__ __align__(16) __bf16 As[128 * LST];
    __shared__ __align__(16) __bf16 Bs[64 * LST];   // transposed: [n][k]

    const int t    = threadIdx.x;
    const int wave = t >> 5;
    const int lane = t & 31;
    const int ln16 = lane & 15;
    const int khalf = (lane >> 4) * 8;
    const int rbase = (lane >> 4) * 8;

    const int m0 = blockIdx.y * 128;
    const int n0 = blockIdx.x * 64;
    const int wm = (wave & 3) * 32;
    const int wn = (wave >> 2) * 32;

    v8f acc[2][2] = {};

    for (int kb = 0; kb < Kdim; kb += 64) {
        // ---- stage A tile: 128 x 64 -> bf16 LDS ----
        if (A_BF16) {
            const __bf16* A = (const __bf16*)Aptr;
            #pragma unroll
            for (int i = 0; i < 4; ++i) {               // 8192 el / (256*8)
                int chunk = t + i * 256;
                int row = chunk >> 3, col = (chunk & 7) * 8;
                *(v8bf*)&As[row * LST + col] =
                    *(const v8bf*)(A + (size_t)(m0 + row) * Kdim + kb + col);
            }
            if (kb + 64 < Kdim)   // prefetch next k-chunk (global_prefetch_b8)
                __builtin_prefetch(A + (size_t)(m0 + (t >> 3)) * Kdim + kb + 64, 0, 3);
        } else {
            const float* A = (const float*)Aptr;
            #pragma unroll
            for (int i = 0; i < 8; ++i) {               // 8192 el / (256*4)
                int chunk = t + i * 256;
                int row = chunk >> 4, col = (chunk & 15) * 4;
                v4f v = *(const v4f*)(A + (size_t)(m0 + row) * Kdim + kb + col);
                __bf16* dst = &As[row * LST + col];
                dst[0] = (__bf16)v.x; dst[1] = (__bf16)v.y;
                dst[2] = (__bf16)v.z; dst[3] = (__bf16)v.w;
            }
            if (kb + 64 < Kdim)
                __builtin_prefetch(A + (size_t)(m0 + (t >> 4)) * Kdim + kb + 64, 0, 3);
        }
        // ---- stage W tile: 64(k) x 64(n) fp32 -> bf16 LDS, transposed ----
        #pragma unroll
        for (int i = 0; i < 4; ++i) {                   // 4096 el / (256*4)
            int chunk = t + i * 256;
            int kr = chunk >> 4, nc = (chunk & 15) * 4;
            v4f v = *(const v4f*)(W + (size_t)(kb + kr) * N + n0 + nc);
            Bs[(nc + 0) * LST + kr] = (__bf16)v.x;
            Bs[(nc + 1) * LST + kr] = (__bf16)v.y;
            Bs[(nc + 2) * LST + kr] = (__bf16)v.z;
            Bs[(nc + 3) * LST + kr] = (__bf16)v.w;
        }
        if (kb + 64 < Kdim)
            __builtin_prefetch(W + (size_t)(kb + 64 + (t >> 4)) * N + n0, 0, 3);
        __syncthreads();

        // ---- 2 WMMA k-steps over the 64-wide chunk ----
        #pragma unroll
        for (int kk = 0; kk < 64; kk += 32) {
            v16bf af[2], bf[2];
            #pragma unroll
            for (int i = 0; i < 2; ++i)
                af[i] = load_frag(&As[(wm + i*16 + ln16) * LST + kk + khalf]);
            #pragma unroll
            for (int j = 0; j < 2; ++j)
                bf[j] = load_frag(&Bs[(wn + j*16 + ln16) * LST + kk + khalf]);
            #pragma unroll
            for (int i = 0; i < 2; ++i)
                #pragma unroll
                for (int j = 0; j < 2; ++j)
                    acc[i][j] = wmma_bf16(af[i], bf[j], acc[i][j]);
        }
        __syncthreads();
    }

    // ---- epilogue: bias + store ----
    #pragma unroll
    for (int i = 0; i < 2; ++i) {
        #pragma unroll
        for (int j = 0; j < 2; ++j) {
            int col = n0 + wn + j * 16 + ln16;
            float bv = bias[col];
            #pragma unroll
            for (int vr = 0; vr < 8; ++vr) {
                int row = m0 + wm + i * 16 + rbase + vr;
                float val = acc[i][j][vr] + bv;
                if (OUT_MODE == 0) {
                    ((float*)Out)[(size_t)row * N + col] = val;
                } else if (OUT_MODE == 1) {
                    // [B][H][L][DK]: row = b*L + l, col = h*DK + d
                    int b = row >> 11, l = row & (SEQ - 1);
                    int h = col >> 6, d = col & (DK - 1);
                    ((__bf16*)Out)[(((size_t)(b * NHEADS + h) * SEQ) + l) * DK + d] =
                        (__bf16)val;
                } else {
                    // [B][H][DK][L] (V transposed so attention's PV B-tile is
                    // a plain strided 2D tile for the TDM)
                    int b = row >> 11, l = row & (SEQ - 1);
                    int h = col >> 6, d = col & (DK - 1);
                    ((__bf16*)Out)[(((size_t)(b * NHEADS + h) * DK) + d) * SEQ + l] =
                        (__bf16)val;
                }
            }
        }
    }
}

// ---------------------------------------------------------------------------
// Flash attention (causal), one (b, h, 128-query tile) per 256-thread block.
// 8 waves; wave w owns query rows [q0+16w, q0+16w+16).  K [key][dk] and
// V^T [d][key] tiles are staged into double-buffered padded LDS by the TDM:
// wave0 issues tile i+1's DMAs before computing tile i, then waits
// TENSORcnt<=2 (in-order completion => tile i has landed).  Waves whose
// whole 16-row stripe is above the diagonal skip the key tile entirely.
// ---------------------------------------------------------------------------
__global__ __launch_bounds__(256)
void attn_kernel(const __bf16* __restrict__ Q,
                 const __bf16* __restrict__ K,
                 const __bf16* __restrict__ Vt_g,
                 __bf16* __restrict__ Ctx) {
    constexpr int LST = 72;
    __shared__ __align__(16) __bf16 Ks[2][64 * LST];     // [buf][key][dk]
    __shared__ __align__(16) __bf16 Vt[2][64 * LST];     // [buf][d][key]
    __shared__ __align__(16) __bf16 Ps[8][16 * LST];     // per-wave P tile

    const int t    = threadIdx.x;
    const int wave = t >> 5;
    const int lane = t & 31;
    const int ln16 = lane & 15;
    const int khalf = (lane >> 4) * 8;
    const int rbase = (lane >> 4) * 8;

    const int q0 = blockIdx.x * 128;
    const int h  = blockIdx.y;
    const int b  = blockIdx.z;
    const size_t headOff = (size_t)(b * NHEADS + h) * SEQ * DK;
    const __bf16* Qh  = Q    + headOff;
    const __bf16* Kh  = K    + headOff;
    const __bf16* Vth = Vt_g + headOff;      // [DK][SEQ]

    // Q A-fragments for this wave's 16 rows, kept in registers for all tiles.
    const int qrow = q0 + wave * 16 + ln16;
    v16bf qfrag[2];
    #pragma unroll
    for (int kb = 0; kb < 2; ++kb)
        qfrag[kb] = load_frag(Qh + (size_t)qrow * DK + kb * 32 + khalf);

    v8f o[4] = {};
    float mrow[8], lrow[8];
    #pragma unroll
    for (int vr = 0; vr < 8; ++vr) { mrow[vr] = -INFINITY; lrow[vr] = 0.0f; }

    // softmax in exp2 domain: scores * (1/sqrt(dk)) * log2(e)
    const float sc = 0.125f * 1.44269504088896f;

    const unsigned int ksAddr[2] = {
        (unsigned int)(uintptr_t)(void*)&Ks[0][0],
        (unsigned int)(uintptr_t)(void*)&Ks[1][0] };
    const unsigned int vtAddr[2] = {
        (unsigned int)(uintptr_t)(void*)&Vt[0][0],
        (unsigned int)(uintptr_t)(void*)&Vt[1][0] };
    const int wuni  = __builtin_amdgcn_readfirstlane(wave);
    const int qmaxw = __builtin_amdgcn_readfirstlane(q0 + wave * 16 + 15);
    const int ntiles = (q0 >> 6) + 2;        // keys 0 .. q0+127

    // prologue: stage tile 0 into buffer 0
    if (wuni == 0) {
        tdm_load_tile64(ksAddr[0], Kh, DK, SEQ, DK);
        tdm_load_tile64(vtAddr[0], Vth, SEQ, DK, SEQ);
    }

    for (int it = 0; it < ntiles; ++it) {
        const int kt0 = it * 64;
        const int cur = it & 1;
        if (wuni == 0) {
            if (it + 1 < ntiles) {
                const int nx = kt0 + 64;
                tdm_load_tile64(ksAddr[cur ^ 1], Kh + (size_t)nx * DK, DK, SEQ, DK);
                tdm_load_tile64(vtAddr[cur ^ 1], Vth + nx, SEQ, DK, SEQ);
                __builtin_amdgcn_s_wait_tensorcnt(2);   // tile `it` has landed
            } else {
                __builtin_amdgcn_s_wait_tensorcnt(0);
            }
        }
        __syncthreads();

        if (kt0 <= qmaxw) {   // whole-wave diagonal skip (scalar branch)
            // ---- scores: S(16x64) = Q(16x64) * K^T ----
            v8f s[4];
            #pragma unroll
            for (int nt = 0; nt < 4; ++nt) {
                v8f a = {};
                #pragma unroll
                for (int kb = 0; kb < 2; ++kb) {
                    v16bf kf = load_frag(&Ks[cur][(nt * 16 + ln16) * LST + kb * 32 + khalf]);
                    a = wmma_bf16(qfrag[kb], kf, a);
                }
                s[nt] = a;
            }

            // ---- causal mask + online softmax (DPP16 row reductions) ----
            float newm[8];
            #pragma unroll
            for (int vr = 0; vr < 8; ++vr) {
                int q = q0 + wave * 16 + rbase + vr;
                float tmax = -INFINITY;
                #pragma unroll
                for (int nt = 0; nt < 4; ++nt) {
                    int j = kt0 + nt * 16 + ln16;
                    float val = (j <= q) ? s[nt][vr] * sc : -INFINITY;
                    s[nt][vr] = val;
                    tmax = fmaxf(tmax, val);
                }
                newm[vr] = fmaxf(mrow[vr], row16_max(tmax));
            }
            #pragma unroll
            for (int vr = 0; vr < 8; ++vr) {
                float f = exp2f(mrow[vr] - newm[vr]);
                mrow[vr] = newm[vr];
                float rsum = 0.0f;
                #pragma unroll
                for (int nt = 0; nt < 4; ++nt) {
                    float p = exp2f(s[nt][vr] - newm[vr]);
                    s[nt][vr] = p;
                    rsum += p;
                }
                lrow[vr] = lrow[vr] * f + row16_sum(rsum);
                #pragma unroll
                for (int nt = 0; nt < 4; ++nt)
                    o[nt][vr] *= f;
            }

            // ---- P (C-layout) -> wave-private LDS -> A-layout fragments ----
            #pragma unroll
            for (int nt = 0; nt < 4; ++nt)
                #pragma unroll
                for (int vr = 0; vr < 8; ++vr)
                    Ps[wave][(rbase + vr) * LST + nt * 16 + ln16] = (__bf16)s[nt][vr];
            // wave-private buffer: raw DS-counter wait instead of a block barrier
            asm volatile("s_wait_dscnt 0x0" ::: "memory");

            // ---- O(16x64) += P(16x64) * V(64x64) ----
            #pragma unroll
            for (int kb = 0; kb < 2; ++kb) {
                v16bf pfrag = load_frag(&Ps[wave][ln16 * LST + kb * 32 + khalf]);
                #pragma unroll
                for (int nt = 0; nt < 4; ++nt) {
                    v16bf vfrag = load_frag(&Vt[cur][(nt * 16 + ln16) * LST + kb * 32 + khalf]);
                    o[nt] = wmma_bf16(pfrag, vfrag, o[nt]);
                }
            }
        }
        __syncthreads();   // all waves done with buf `cur` before it is re-DMAed
    }

    // ---- normalize + store context bf16 in [B, L, H*DK] ----
    #pragma unroll
    for (int vr = 0; vr < 8; ++vr) {
        float inv = 1.0f / lrow[vr];
        int q = q0 + wave * 16 + rbase + vr;
        size_t rowOff = ((size_t)b * SEQ + q) * DMODEL + (size_t)h * DK;
        #pragma unroll
        for (int nt = 0; nt < 4; ++nt)
            Ctx[rowOff + nt * 16 + ln16] = (__bf16)(o[nt][vr] * inv);
    }
}

// ---------------------------------------------------------------------------
// Host launcher.  Inputs: x, mask, Wq, bq, Wk, bk, Wv, bv, Wo, bo.
// Workspace: Qb | Kb | Vb^T | Ctx, all bf16, 16 MB each (64 MB total).
// ---------------------------------------------------------------------------
extern "C" void kernel_launch(void* const* d_in, const int* in_sizes, int n_in,
                              void* d_out, int out_size, void* d_ws, size_t ws_size,
                              hipStream_t stream) {
    const float* x  = (const float*)d_in[0];
    // d_in[1] is the causal mask tensor: computed analytically, not read.
    const float* Wq = (const float*)d_in[2];
    const float* bq = (const float*)d_in[3];
    const float* Wk = (const float*)d_in[4];
    const float* bk = (const float*)d_in[5];
    const float* Wv = (const float*)d_in[6];
    const float* bv = (const float*)d_in[7];
    const float* Wo = (const float*)d_in[8];
    const float* bo = (const float*)d_in[9];

    const size_t headElems = (size_t)BATCH * NHEADS * SEQ * DK;   // 8,388,608
    char* ws = (char*)d_ws;
    __bf16* Qb  = (__bf16*)(ws);
    __bf16* Kb  = (__bf16*)(ws + 2 * headElems);
    __bf16* Vbt = (__bf16*)(ws + 4 * headElems);
    __bf16* Ctx = (__bf16*)(ws + 6 * headElems);

    dim3 gemmGrid(DMODEL / 64, MROWS / 128);   // (16, 64)
    dim3 gemmBlk(256);

    // Q/K projections -> bf16 [B,H,L,DK]; V projection -> bf16 [B,H,DK,L]
    gemm_wmma_kernel<false, 1><<<gemmGrid, gemmBlk, 0, stream>>>(
        x, Wq, bq, Qb, MROWS, DMODEL, DMODEL);
    gemm_wmma_kernel<false, 1><<<gemmGrid, gemmBlk, 0, stream>>>(
        x, Wk, bk, Kb, MROWS, DMODEL, DMODEL);
    gemm_wmma_kernel<false, 2><<<gemmGrid, gemmBlk, 0, stream>>>(
        x, Wv, bv, Vbt, MROWS, DMODEL, DMODEL);

    // causal flash attention (double-buffered TDM staging)
    dim3 attnGrid(SEQ / 128, NHEADS, BATCH);   // (16, 16, 4)
    attn_kernel<<<attnGrid, dim3(256), 0, stream>>>(Qb, Kb, Vbt, Ctx);

    // output projection: bf16 ctx -> fp32 d_out
    gemm_wmma_kernel<true, 0><<<gemmGrid, gemmBlk, 0, stream>>>(
        Ctx, Wo, bo, (float*)d_out, MROWS, DMODEL, DMODEL);
}
